// wtGNN_65549790871882
// MI455X (gfx1250) — matrix operand order repaired and checked by reading, not compiled, not run on hardware
//
#include <hip/hip_runtime.h>
#include <cstdint>

typedef __attribute__((ext_vector_type(16))) __bf16 v16bf;
typedef __attribute__((ext_vector_type(8)))  __bf16 v8bf;
typedef __attribute__((ext_vector_type(8)))  float  v8f;

#define DCH 128   // channels
#define NH  4     // heads
#define CH  32    // channels per head
#define EPW 4     // edges per wave in gather kernels

__device__ inline __bf16 to_bf16(float f) {
    unsigned u = __builtin_bit_cast(unsigned, f);
    // round-to-nearest-even bf16
    unsigned short h = (unsigned short)((u + 0x7FFFu + ((u >> 16) & 1u)) >> 16);
    return __builtin_bit_cast(__bf16, h);
}

__device__ inline void atomicMaxF(float* addr, float val) {
    unsigned* ua = (unsigned*)addr;
    unsigned old = *ua;
    while (__builtin_bit_cast(float, old) < val) {
        unsigned assumed = old;
        old = atomicCAS(ua, assumed, __builtin_bit_cast(unsigned, val));
        if (old == assumed) break;
    }
}

// ---------------------------------------------------------------------------
// Pack Wl/Wr (f32 [128][128], row-major K x N) into bf16 WMMA B-fragment
// layout: wpack[(((mat*8 + tile)*4 + ks)*32 + lane)*16 + j], so each lane of
// the GEMM loads one contiguous 32B v16bf per K-step.  contiguous k order used here (same as A), 32x16 bf16:
// col N = lane&15; lanes 0-15 hold K=0..15, lanes 16-31 hold K=16..31.
// ---------------------------------------------------------------------------
__global__ void gatv2_pack_w(const float* __restrict__ Wl,
                             const float* __restrict__ Wr,
                             __bf16* __restrict__ wpack)
{
    int t = blockIdx.x * blockDim.x + threadIdx.x;   // (mat,tile,ks,lane)
    if (t >= 2 * 8 * 4 * 32) return;
    int lane = t & 31, ks = (t >> 5) & 3, tile = (t >> 7) & 7, mat = t >> 10;
    const float* W = mat ? Wr : Wl;
    int lhalf = lane >> 4, l15 = lane & 15;
    #pragma unroll
    for (int j = 0; j < 16; ++j) {
        int kb = ks * 32 + j + 16 * lhalf;
        wpack[(size_t)t * 16 + j] = to_bf16(W[(size_t)kb * DCH + tile * 16 + l15]);
    }
}

// ---------------------------------------------------------------------------
// Convert f32 features -> bf16 copy (layer 0 only)
// ---------------------------------------------------------------------------
__global__ void gatv2_cvt_x(const float* __restrict__ x,
                            __bf16* __restrict__ xbf, int total)
{
    int i = blockIdx.x * blockDim.x + threadIdx.x;
    if (i < total) xbf[i] = to_bf16(x[i]);
}

// ---------------------------------------------------------------------------
// Fused GEMM: xl = x@Wl, xr = x@Wr from bf16 x and packed bf16 weights.
// grid.x = ceil(n/16) row-slabs; block = 256 = 8 waves; wave w owns output
// cols [16w,16w+16) for BOTH outputs (A fragment reused -> 8 WMMAs/wave).
// A-slab staged into LDS with GLOBAL_LOAD_ASYNC_TO_LDS_B128 (ASYNCcnt path).
// ---------------------------------------------------------------------------
__global__ void gatv2_gemm_xw(const __bf16* __restrict__ xbf,
                              const __bf16* __restrict__ wpack,
                              float* __restrict__ xl,
                              float* __restrict__ xr,
                              int n)
{
    __shared__ __bf16 As[16 * DCH];   // 16 rows x 128 K (bf16), 4KB

    const int row0 = blockIdx.x * 16;
    const bool full = (row0 + 16 <= n);

    if (full) {
        // async copy: each of 256 threads moves 16B global -> LDS
        unsigned lds = (unsigned)(uintptr_t)(const void*)As + threadIdx.x * 16u;
        const __bf16* g = xbf + (size_t)row0 * DCH + threadIdx.x * 8;
        asm volatile("global_load_async_to_lds_b128 %0, %1, off"
                     :: "v"(lds), "v"(g) : "memory");
        asm volatile("s_wait_asynccnt 0x0" ::: "memory");
    } else {
        for (int i = threadIdx.x; i < 16 * DCH; i += 256) {
            int r = i >> 7, k = i & (DCH - 1);
            As[i] = (row0 + r < n) ? xbf[(size_t)(row0 + r) * DCH + k]
                                   : __builtin_bit_cast(__bf16, (unsigned short)0);
        }
    }
    __syncthreads();

    const int wave  = threadIdx.x >> 5;   // 0..7 -> column tile
    const int lane  = threadIdx.x & 31;
    const int lhalf = lane >> 4;
    const int l15   = lane & 15;
    const int ncol0 = wave * 16;

    const v16bf* WP = (const v16bf*)wpack;

    v8f accl = {}, accr = {};
    #pragma unroll
    for (int ks = 0; ks < 4; ++ks) {
        const int kk = ks * 32;
        // A (16x32 bf16): row = l15; K = 0..15 (lanes 0-15), 16..31 (lanes 16-31)
        v8bf a0 = *(const v8bf*)&As[l15 * DCH + kk + 16 * lhalf];
        v8bf a1 = *(const v8bf*)&As[l15 * DCH + kk + 16 * lhalf + 8];
        v16bf a;
        #pragma unroll
        for (int j = 0; j < 8; ++j) { a[j] = a0[j]; a[8 + j] = a1[j]; }

        v16bf bl = WP[((size_t)(0 * 8 + wave) * 4 + ks) * 32 + lane];
        v16bf br = WP[((size_t)(1 * 8 + wave) * 4 + ks) * 32 + lane];

        accl = __builtin_amdgcn_wmma_f32_16x16x32_bf16(false, a, false, bl,
                                                       (short)0, accl, false, false);
        accr = __builtin_amdgcn_wmma_f32_16x16x32_bf16(false, a, false, br,
                                                       (short)0, accr, false, false);
    }

    // C/D layout: VGPR r -> M = r + 8*(lane>=16); N = l15
    float* pl = xl + (size_t)(row0 + 8 * lhalf) * DCH + ncol0 + l15;
    float* pr = xr + (size_t)(row0 + 8 * lhalf) * DCH + ncol0 + l15;
    if (full) {                       // branch-free common path
        #pragma unroll
        for (int r = 0; r < 8; ++r) {
            pl[(size_t)r * DCH] = accl[r];
            pr[(size_t)r * DCH] = accr[r];
        }
    } else {
        #pragma unroll
        for (int r = 0; r < 8; ++r) {
            int m = r + 8 * lhalf;
            if (row0 + m < n) {
                pl[(size_t)r * DCH] = accl[r];
                pr[(size_t)r * DCH] = accr[r];
            }
        }
    }
}

// ---------------------------------------------------------------------------
// Per-layer init: agg = 0, emax = -inf, denom = 0
// ---------------------------------------------------------------------------
__global__ void gatv2_init(float* __restrict__ agg,
                           float* __restrict__ emax,
                           float* __restrict__ denom,
                           int n)
{
    int i = blockIdx.x * blockDim.x + threadIdx.x;
    if (i < n * DCH) agg[i] = 0.0f;
    if (i < n * NH) { emax[i] = -3.4e38f; denom[i] = 0.0f; }
}

// ---------------------------------------------------------------------------
// Pass 1: e = att . leakyrelu(xl[src] + xr[dst]); segment atomic max.
// 4 edges per wave with up-front row prefetch (lane-strided, covers 512B row).
// Lane l covers channel l of each head -> shuffle reduce gives all 4 logits.
// ---------------------------------------------------------------------------
__global__ void gatv2_edge_logits(const long long* __restrict__ ei, int E_, int ET,
                                  const float* __restrict__ xl,
                                  const float* __restrict__ xr,
                                  const float* __restrict__ att,   // [4][32]
                                  float* __restrict__ ebuf,        // [ET][4]
                                  float* __restrict__ emax)        // [n][4]
{
    const int wv   = blockIdx.x * 8 + (threadIdx.x >> 5);
    const int lane = threadIdx.x & 31;
    const int base = wv * EPW;

    long long ss[EPW], dd[EPW];
    #pragma unroll
    for (int q = 0; q < EPW; ++q) {
        int edge = base + q;
        if (edge >= ET) { ss[q] = dd[q] = 0; continue; }
        if (edge < E_) { ss[q] = ei[edge]; dd[q] = ei[E_ + edge]; }
        else           { ss[q] = dd[q] = (long long)(edge - E_); }
        __builtin_prefetch(xl + (size_t)ss[q] * DCH + lane * 4, 0, 0);
        __builtin_prefetch(xr + (size_t)dd[q] * DCH + lane * 4, 0, 0);
    }

    const float av0 = att[0 * CH + lane], av1 = att[1 * CH + lane];
    const float av2 = att[2 * CH + lane], av3 = att[3 * CH + lane];

    #pragma unroll
    for (int q = 0; q < EPW; ++q) {
        int edge = base + q;
        if (edge >= ET) break;
        const float* pxl = xl + (size_t)ss[q] * DCH;
        const float* pxr = xr + (size_t)dd[q] * DCH;

        float part[NH];
        #pragma unroll
        for (int h = 0; h < NH; ++h) {
            float v = pxl[h * CH + lane] + pxr[h * CH + lane];
            v = (v > 0.0f) ? v : 0.2f * v;                 // leaky relu 0.2
            float a = (h == 0) ? av0 : (h == 1) ? av1 : (h == 2) ? av2 : av3;
            part[h] = v * a;
        }
        #pragma unroll
        for (int off = 16; off > 0; off >>= 1) {
            #pragma unroll
            for (int h = 0; h < NH; ++h)
                part[h] += __shfl_xor(part[h], off, 32);
        }
        if (lane < NH) {
            float e = part[lane];
            ebuf[(size_t)edge * NH + lane] = e;
            atomicMaxF(&emax[(size_t)dd[q] * NH + lane], e);
        }
    }
}

// ---------------------------------------------------------------------------
// Pass 2: ex = exp(e - emax[dst]); denom[dst] += ex.  Thread per (edge,head).
// ---------------------------------------------------------------------------
__global__ void gatv2_edge_exp(const long long* __restrict__ ei, int E_, int ET,
                               float* __restrict__ ebuf,
                               const float* __restrict__ emax,
                               float* __restrict__ denom)
{
    int idx = blockIdx.x * blockDim.x + threadIdx.x;   // edge*4 + h
    if (idx >= ET * NH) return;
    int edge = idx >> 2, h = idx & 3;
    long long d = (edge < E_) ? ei[E_ + edge] : (long long)(edge - E_);
    float ex = __expf(ebuf[idx] - emax[(size_t)d * NH + h]);
    ebuf[idx] = ex;
    atomicAdd(&denom[(size_t)d * NH + h], ex);
}

// ---------------------------------------------------------------------------
// Pass 3: agg[dst] += alpha * xl[src].  4 edges per wave + row prefetch;
// lane l covers channel l of each head -> 4 float atomics per lane per edge.
// ---------------------------------------------------------------------------
__global__ void gatv2_edge_agg(const long long* __restrict__ ei, int E_, int ET,
                               const float* __restrict__ xl,
                               const float* __restrict__ ebuf,
                               const float* __restrict__ denom,
                               float* __restrict__ agg)
{
    const int wv   = blockIdx.x * 8 + (threadIdx.x >> 5);
    const int lane = threadIdx.x & 31;
    const int base = wv * EPW;

    long long ss[EPW], dd[EPW];
    #pragma unroll
    for (int q = 0; q < EPW; ++q) {
        int edge = base + q;
        if (edge >= ET) { ss[q] = dd[q] = 0; continue; }
        if (edge < E_) { ss[q] = ei[edge]; dd[q] = ei[E_ + edge]; }
        else           { ss[q] = dd[q] = (long long)(edge - E_); }
        __builtin_prefetch(xl + (size_t)ss[q] * DCH + lane * 4, 0, 0);
    }

    #pragma unroll
    for (int q = 0; q < EPW; ++q) {
        int edge = base + q;
        if (edge >= ET) break;
        #pragma unroll
        for (int h = 0; h < NH; ++h) {
            float alpha = ebuf[(size_t)edge * NH + h] /
                          (denom[(size_t)dd[q] * NH + h] + 1e-16f);
            atomicAdd(&agg[(size_t)dd[q] * DCH + h * CH + lane],
                      alpha * xl[(size_t)ss[q] * DCH + h * CH + lane]);
        }
    }
}

// ---------------------------------------------------------------------------
// Finalize: v = relu(agg + bias); write f32 layer output + bf16 copy for the
// next layer's WMMA GEMM.
// ---------------------------------------------------------------------------
__global__ void gatv2_finalize(const float* __restrict__ agg,
                               const float* __restrict__ bias,
                               float* __restrict__ out,
                               __bf16* __restrict__ xbf,
                               int total)
{
    int i = blockIdx.x * blockDim.x + threadIdx.x;
    if (i >= total) return;
    float v = agg[i] + bias[i & (DCH - 1)];
    v = (v > 0.0f) ? v : 0.0f;
    out[i] = v;
    xbf[i] = to_bf16(v);
}

// ---------------------------------------------------------------------------
extern "C" void kernel_launch(void* const* d_in, const int* in_sizes, int n_in,
                              void* d_out, int out_size, void* d_ws, size_t ws_size,
                              hipStream_t stream)
{
    const float*     x0 = (const float*)d_in[0];       // [1,N,128] f32
    const long long* ei = (const long long*)d_in[1];   // [2,E] int64

    const int n  = in_sizes[0] / DCH;   // 100000
    const int E_ = in_sizes[1] / 2;     // 600000
    const int ET = E_ + n;              // + self loops

    // workspace layout
    float*  bufA  = (float*)d_ws;                 // agg / layer output, n*128
    float*  bufB  = bufA + (size_t)n * DCH;       // xl
    float*  bufC  = bufB + (size_t)n * DCH;       // xr
    float*  ebuf  = bufC + (size_t)n * DCH;       // ET*4
    float*  emax  = ebuf + (size_t)ET * NH;       // n*4
    float*  denom = emax + (size_t)n * NH;        // n*4
    __bf16* xbf   = (__bf16*)(denom + (size_t)n * NH);        // n*128 bf16
    __bf16* wpack = xbf + (size_t)n * DCH;                    // 2*128*128 bf16

    const dim3 gemmGrid((n + 15) / 16);
    const int  initBlocks = (n * DCH + 255) / 256;
    const int  edgeBlocks = (ET + 8 * EPW - 1) / (8 * EPW);
    const int  expBlocks  = (ET * NH + 255) / 256;
    const int  finBlocks  = (n * DCH + 255) / 256;

    gatv2_cvt_x<<<finBlocks, 256, 0, stream>>>(x0, xbf, n * DCH);

    for (int layer = 0; layer < 4; ++layer) {
        const float* Wl  = (const float*)d_in[2 + 4 * layer + 0];
        const float* Wr  = (const float*)d_in[2 + 4 * layer + 1];
        const float* att = (const float*)d_in[2 + 4 * layer + 2];
        const float* bia = (const float*)d_in[2 + 4 * layer + 3];

        gatv2_pack_w<<<8, 256, 0, stream>>>(Wl, Wr, wpack);
        gatv2_gemm_xw<<<gemmGrid, 256, 0, stream>>>(xbf, wpack, bufB, bufC, n);
        // agg reuses bufA -- safe: GEMM consumed x (bf16 copy) already
        gatv2_init<<<initBlocks, 256, 0, stream>>>(bufA, emax, denom, n);
        gatv2_edge_logits<<<edgeBlocks, 256, 0, stream>>>(ei, E_, ET, bufB, bufC,
                                                          att, ebuf, emax);
        gatv2_edge_exp<<<expBlocks, 256, 0, stream>>>(ei, E_, ET, ebuf, emax, denom);
        gatv2_edge_agg<<<edgeBlocks, 256, 0, stream>>>(ei, E_, ET, bufB, ebuf,
                                                       denom, bufA);
        float* outp = (layer == 3) ? (float*)d_out : bufA;
        gatv2_finalize<<<finBlocks, 256, 0, stream>>>(bufA, bia, outp, xbf, n * DCH);
    }
}
